// MultiQueryAtt_16372415332396
// MI455X (gfx1250) — matrix operand order repaired and checked by reading, not compiled, not run on hardware
//
#include <hip/hip_runtime.h>
#include <hip/hip_bf16.h>

typedef __attribute__((ext_vector_type(16))) _Float16 v16h;
typedef __attribute__((ext_vector_type(8)))  float    v8f;

#define L_SEQ   2048
#define D_HEAD  64
#define KTILES  (L_SEQ / 16)     // 128 k-tiles per row block
#define WAVES   4
#define THREADS (WAVES * 32)

// Fused masked attention: one workgroup = one (b*h, 16-row q tile).
// LDS holds the full 16 x 2048 score row block (128 KB) so softmax is exact.
__global__ __launch_bounds__(THREADS)
void mqa_fused_kernel(const float* __restrict__ q,
                      const float* __restrict__ k,
                      const float* __restrict__ v,
                      const int*   __restrict__ valid,
                      const float* __restrict__ scale_p,
                      float* __restrict__ out_res,
                      float* __restrict__ out_attw)
{
    extern __shared__ float smem[];
    float* sc   = smem;                 // [16][L_SEQ] scores -> probs
    float* red  = smem + 16 * L_SEQ;    // [THREADS] reduction scratch
    float* rmax = red + THREADS;        // [16] row max
    float* rinv = rmax + 16;            // [16] 1/rowsum (0 if fully masked)

    const int qt   = blockIdx.x;        // 0..127 q tile
    const int bh   = blockIdx.y;        // 0..31  fused batch*head
    const int tid  = threadIdx.x;
    const int wave = tid >> 5;
    const int lane = tid & 31;
    const float scale = scale_p[0];

    const size_t head_base = (size_t)bh * L_SEQ * D_HEAD;  // q/k/v head offset
    const int qrow0 = qt * 16;

    // ---- Q A-fragments: 16x64 as two 16x32 f16 fragments -------------------
    // A layout (ISA 7.12.2, 16-bit 16x32): lane holds row (lane&15);
    // lanes 0-15: K {0..7,16..23}, lanes 16-31: K {8..15,24..31}.
    const int am   = lane & 15;
    const int asel = (lane >> 4) * 8;
    v16h a0, a1;
    {
        const float* qr = q + head_base + (size_t)(qrow0 + am) * D_HEAD;
        #pragma unroll
        for (int i = 0; i < 8; ++i) {
            a0[i]     = (_Float16)qr[ 0 + asel + i];
            a0[8 + i] = (_Float16)qr[16 + asel + i];
            a1[i]     = (_Float16)qr[32 + asel + i];
            a1[8 + i] = (_Float16)qr[48 + asel + i];
        }
    }

    // ---- Stage 1: S = scale * Q K^T, masked to -inf, into LDS --------------
    // B layout (32x16): lane holds column (lane&15) = K-matrix row (row-major),
    // lanes 0-15 inner-k {0..7,16..23}, lanes 16-31 inner-k {8..15,24..31}.
    const int bn    = lane & 15;
    const int bhalf = lane >> 4;
    const size_t mrow_base = ((size_t)bh * L_SEQ + qrow0) * L_SEQ;  // valid/att_w
    for (int kt = wave; kt < KTILES; kt += WAVES) {
        const float* kr = k + head_base + (size_t)(kt * 16 + bn) * D_HEAD;
        v16h b0, b1;
        #pragma unroll
        for (int i = 0; i < 8; ++i) {
            b0[i]     = (_Float16)kr[     bhalf * 8 + i];   // inner K 0..31
            b0[8 + i] = (_Float16)kr[16 + bhalf * 8 + i];
            b1[i]     = (_Float16)kr[32 + bhalf * 8 + i];   // inner K 32..63
            b1[8 + i] = (_Float16)kr[48 + bhalf * 8 + i];
        }
        v8f c = {};
        c = __builtin_amdgcn_wmma_f32_16x16x32_f16(false, a0, false, b0, (short)0, c, false, false);
        c = __builtin_amdgcn_wmma_f32_16x16x32_f16(false, a1, false, b1, (short)0, c, false, false);

        // C layout: VGPR r, lane l -> (M = r + 8*(l>>4), N = l&15)
        const int kcol = kt * 16 + (lane & 15);
        #pragma unroll
        for (int r = 0; r < 8; ++r) {
            const int ml = r + 8 * (lane >> 4);
            float s = c[r] * scale;
            const int ok = valid[mrow_base + (size_t)ml * L_SEQ + kcol];
            if (ok <= 0) s = -__builtin_inff();
            sc[ml * L_SEQ + kcol] = s;
        }
    }
    __syncthreads();

    // ---- Stage 2: row softmax stats (max, exp, sum) ------------------------
    {
        const int row = tid >> 3;       // 8 threads per row
        const int sub = tid & 7;
        float m = -__builtin_inff();
        for (int c0 = sub; c0 < L_SEQ; c0 += 8)
            m = fmaxf(m, sc[row * L_SEQ + c0]);
        red[tid] = m;
        __syncthreads();
        if (tid < 16) {
            float mm = red[tid * 8];
            #pragma unroll
            for (int i = 1; i < 8; ++i) mm = fmaxf(mm, red[tid * 8 + i]);
            rmax[tid] = mm;
        }
        __syncthreads();
        const float mrow = rmax[row];
        float ssum = 0.f;
        for (int c0 = sub; c0 < L_SEQ; c0 += 8) {
            const float sv = sc[row * L_SEQ + c0];
            // masked (or fully-masked row) entries contribute exactly 0
            const float e = (sv == -__builtin_inff()) ? 0.f : __expf(sv - mrow);
            sc[row * L_SEQ + c0] = e;
            ssum += e;
        }
        red[tid] = ssum;
        __syncthreads();
        if (tid < 16) {
            float t = 0.f;
            #pragma unroll
            for (int i = 0; i < 8; ++i) t += red[tid * 8 + i];
            rinv[tid] = (t > 0.f) ? 1.0f / t : 0.f;   // NaN->0 path of reference
        }
        __syncthreads();
    }

    // ---- Stage 3: normalize, stream att_w out (coalesced), keep P in LDS ---
    for (int i = tid; i < 16 * L_SEQ; i += THREADS) {
        const int row = i >> 11;                 // L_SEQ == 2048
        const float p = sc[i] * rinv[row];
        sc[i] = p;
        out_attw[mrow_base + (size_t)i] = p;     // row-contiguous block
    }
    __syncthreads();

    // ---- Stage 4: O = P V. Each wave owns one 16-wide n-tile of D ----------
    {
        v8f acc = {};
        const int n0 = wave * 16;
        for (int ch = 0; ch < L_SEQ / 32; ++ch) {
            v16h pa, vb;
            const float* prow = sc + am * L_SEQ + ch * 32;
            #pragma unroll
            for (int i = 0; i < 8; ++i) {
                pa[i]     = (_Float16)prow[asel + i];
                pa[8 + i] = (_Float16)prow[16 + asel + i];
            }
            // column n0+bn of V for inner rows ch*32 + bhalf*8 + {i, 16+i}
            const float* vcol = v + head_base
                              + (size_t)(ch * 32 + bhalf * 8) * D_HEAD + n0 + bn;
            #pragma unroll
            for (int i = 0; i < 8; ++i) {
                vb[i]     = (_Float16)vcol[(size_t)i * D_HEAD];
                vb[8 + i] = (_Float16)vcol[(size_t)(16 + i) * D_HEAD];
            }
            acc = __builtin_amdgcn_wmma_f32_16x16x32_f16(false, pa, false, vb, (short)0, acc, false, false);
        }
        const size_t obase = ((size_t)bh * L_SEQ + qrow0) * D_HEAD;
        #pragma unroll
        for (int r = 0; r < 8; ++r) {
            const int ml = r + 8 * (lane >> 4);
            out_res[obase + (size_t)ml * D_HEAD + n0 + bn] = acc[r];
        }
    }
}

extern "C" void kernel_launch(void* const* d_in, const int* in_sizes, int n_in,
                              void* d_out, int out_size, void* d_ws, size_t ws_size,
                              hipStream_t stream) {
    (void)in_sizes; (void)n_in; (void)out_size; (void)d_ws; (void)ws_size;
    const float* q     = (const float*)d_in[0];
    const float* k     = (const float*)d_in[1];
    const float* v     = (const float*)d_in[2];
    const int*   valid = (const int*)  d_in[3];
    const float* scale = (const float*)d_in[4];

    float* out_res  = (float*)d_out;                       // [2,16,2048,64]
    float* out_attw = out_res + (size_t)2 * 16 * 2048 * 64; // [2,16,2048,2048]

    dim3 grid(L_SEQ / 16, 2 * 16);   // (q tiles, B*H)
    dim3 block(THREADS);
    size_t shmem = (size_t)(16 * L_SEQ + THREADS + 16 + 16) * sizeof(float);
    mqa_fused_kernel<<<grid, block, shmem, stream>>>(q, k, v, valid, scale,
                                                     out_res, out_attw);
}
